// SmMPNN_22273700397415
// MI455X (gfx1250) — matrix operand order repaired and checked by reading, not compiled, not run on hardware
//
#include <hip/hip_runtime.h>
#include <hip/hip_bf16.h>

#define HIDDEN 64
#define NGRAPHS 64

typedef __attribute__((ext_vector_type(16))) __bf16 v16bf;
typedef __attribute__((ext_vector_type(8)))  float  v8f;

// ---------------- workspace layout (floats) ----------------
// [0,64)        A
// [64,128)      B
// [128,192)     C
// [192,256)     counts (G=64)
// [256,4352)    sums   (G=64 x H=64)
// [4352, ...)   aggr   (N x 64)
#define WS_A      0
#define WS_B      64
#define WS_C      128
#define WS_CNT    192
#define WS_SUMS   256
#define WS_AGGR   4352

// ------------------------------------------------------------------
// Zero the scratch region we accumulate into.
// ------------------------------------------------------------------
__global__ void zero_kernel(float* ws, long long nTotal) {
    long long i = (long long)blockIdx.x * blockDim.x + threadIdx.x;
    long long stride = (long long)gridDim.x * blockDim.x;
    for (; i < nTotal; i += stride) ws[i] = 0.0f;
}

// ------------------------------------------------------------------
// Fold encoders into the message MLP (exact rank-1 collapse).
// A[k] = sum_j Wn[j]*Wm[j][k];  B[k] = sum_j We[j]*Wm[64+j][k]
// C[k] = sum_j bn[j]*Wm[j][k] + be[j]*Wm[64+j][k] + bm[k]
// ------------------------------------------------------------------
__global__ void fold_kernel(const float* __restrict__ Wn, const float* __restrict__ bn,
                            const float* __restrict__ We, const float* __restrict__ be,
                            const float* __restrict__ Wm, const float* __restrict__ bm,
                            float* __restrict__ ws) {
    int k = threadIdx.x;             // 0..63
    float a = 0.f, b = 0.f, c = bm[k];
    #pragma unroll 8
    for (int j = 0; j < HIDDEN; ++j) {
        float wlo = Wm[j * HIDDEN + k];
        float whi = Wm[(HIDDEN + j) * HIDDEN + k];
        a += Wn[j] * wlo;
        b += We[j] * whi;
        c += bn[j] * wlo + be[j] * whi;
    }
    ws[WS_A + k] = a;
    ws[WS_B + k] = b;
    ws[WS_C + k] = c;
}

// ------------------------------------------------------------------
// Edge phase: m[e,k] = relu(x[src]*A[k] + ea[e]*B[k] + C[k]),
// scatter-added into aggr[dst]. 32 lanes per edge, 2 features each.
// ------------------------------------------------------------------
__global__ __launch_bounds__(256) void edge_kernel(
        const float* __restrict__ x, const float* __restrict__ ea,
        const int* __restrict__ eidx, const float* __restrict__ ws,
        float* __restrict__ aggr, int nEdges) {
    __shared__ float sA[HIDDEN], sB[HIDDEN], sC[HIDDEN];
    int t = threadIdx.x;
    if (t < 64)       sA[t]       = ws[WS_A + t];
    else if (t < 128) sB[t - 64]  = ws[WS_B + t - 64];
    else if (t < 192) sC[t - 128] = ws[WS_C + t - 128];
    __syncthreads();

    long long gid = (long long)blockIdx.x * 256 + t;
    int e    = (int)(gid >> 5);
    int lane = (int)(gid & 31);
    if (e >= nEdges) return;

    int   src = eidx[e];
    int   dst = eidx[nEdges + e];
    float xv  = x[src];
    float ev  = ea[e];

    float* dstp = aggr + (long long)dst * HIDDEN;
    int f0 = lane, f1 = lane + 32;
    float m0 = fmaxf(fmaf(xv, sA[f0], fmaf(ev, sB[f0], sC[f0])), 0.0f);
    float m1 = fmaxf(fmaf(xv, sA[f1], fmaf(ev, sB[f1], sC[f1])), 0.0f);
    unsafeAtomicAdd(dstp + f0, m0);   // global_atomic_add_f32
    unsafeAtomicAdd(dstp + f1, m1);
}

// ------------------------------------------------------------------
// Per-graph node counts.
// ------------------------------------------------------------------
__global__ void count_kernel(const int* __restrict__ batch, float* __restrict__ cnt, int nNodes) {
    int i = blockIdx.x * blockDim.x + threadIdx.x;
    int stride = gridDim.x * blockDim.x;
    for (; i < nNodes; i += stride)
        unsafeAtomicAdd(&cnt[batch[i]], 1.0f);
}

// ------------------------------------------------------------------
// Update MLP via WMMA bf16->f32 + fused ReLU + graph pooling.
// One wave (32 threads) per 16-node tile. D = aggr_tile(16x64) @ Wupd(64x64).
// K split into two 16x16x32 steps; N split into four 16-col tiles.
// A fragment k order used here (contiguous, not ISA table order): slot s -> K = s + 16*hi
// B fragment, same contiguous k order as A: lane half holds K 0..15 / 16..31,
//   slot s -> K = 16*hi + s, N = 16*t + (lane&15).
// D layout: VGPR r -> row = r + 8*hi, col = lane&15.
// ------------------------------------------------------------------
__global__ __launch_bounds__(32) void upd_pool_kernel(
        const float* __restrict__ aggr, const float* __restrict__ Wupd,
        const float* __restrict__ bupd, const int* __restrict__ batch,
        float* __restrict__ sums) {
    int tile = blockIdx.x;               // node tile: 16 rows
    int lane = threadIdx.x;              // 0..31
    int m    = lane & 15;
    int hi   = lane >> 4;
    int rowBase = tile * 16;

    // --- A fragments for K=[0,32) and K=[32,64), converted f32 -> bf16 ---
    const float* arow = aggr + (long long)(rowBase + m) * HIDDEN;
    v16bf a0, a1;
    #pragma unroll
    for (int s = 0; s < 16; ++s) {
        int koff = s + (hi ? 16 : 0);
        a0[s] = (__bf16)arow[koff];
        a1[s] = (__bf16)arow[32 + koff];
    }

    v8f acc[4];
    #pragma unroll
    for (int t = 0; t < 4; ++t) {
        int n  = t * 16 + m;
        int kb = hi * 16;
        v16bf b0, b1;
        #pragma unroll
        for (int s = 0; s < 16; ++s) {
            b0[s] = (__bf16)Wupd[(kb + s) * HIDDEN + n];
            b1[s] = (__bf16)Wupd[(32 + kb + s) * HIDDEN + n];
        }
        v8f c = {};
        c = __builtin_amdgcn_wmma_f32_16x16x32_bf16(false, a0, false, b0,
                                                    (short)0, c, false, false);
        c = __builtin_amdgcn_wmma_f32_16x16x32_bf16(false, a1, false, b1,
                                                    (short)0, c, false, false);
        acc[t] = c;
    }

    // --- bias + ReLU + pool directly into per-graph sums ---
    #pragma unroll
    for (int t = 0; t < 4; ++t) {
        int feat = t * 16 + m;
        float bv = bupd[feat];
        #pragma unroll
        for (int r = 0; r < 8; ++r) {
            int row = rowBase + r + hi * 8;
            float v = fmaxf(acc[t][r] + bv, 0.0f);
            int g = batch[row];
            unsafeAtomicAdd(&sums[g * HIDDEN + feat], v);
        }
    }
}

// ------------------------------------------------------------------
// out[g] = (sum_k sums[g,k]*Wout[k]) / max(count[g],1) + bout
// ------------------------------------------------------------------
__global__ void out_kernel(const float* __restrict__ ws, const float* __restrict__ Wout,
                           const float* __restrict__ bout, float* __restrict__ out) {
    int g = threadIdx.x;                 // 0..63
    const float* srow = ws + WS_SUMS + g * HIDDEN;
    float acc = 0.f;
    #pragma unroll 8
    for (int k = 0; k < HIDDEN; ++k) acc += srow[k] * Wout[k];
    float cnt = fmaxf(ws[WS_CNT + g], 1.0f);
    out[g] = acc / cnt + bout[0];
}

// ------------------------------------------------------------------
extern "C" void kernel_launch(void* const* d_in, const int* in_sizes, int n_in,
                              void* d_out, int out_size, void* d_ws, size_t ws_size,
                              hipStream_t stream) {
    const float* x    = (const float*)d_in[0];
    const float* ea   = (const float*)d_in[1];
    const int*   eidx = (const int*)  d_in[2];
    const int*   batc = (const int*)  d_in[3];
    const float* Wn   = (const float*)d_in[4];
    const float* bn   = (const float*)d_in[5];
    const float* We   = (const float*)d_in[6];
    const float* be   = (const float*)d_in[7];
    const float* Wm   = (const float*)d_in[8];
    const float* bm   = (const float*)d_in[9];
    const float* Wupd = (const float*)d_in[10];
    const float* bupd = (const float*)d_in[11];
    const float* Wout = (const float*)d_in[12];
    const float* bout = (const float*)d_in[13];
    float* out = (float*)d_out;
    float* ws  = (float*)d_ws;

    int nNodes = in_sizes[0];            // 50000
    int nEdges = in_sizes[1];            // 800000
    float* aggr = ws + WS_AGGR;

    long long zTotal = (long long)WS_AGGR + (long long)nNodes * HIDDEN;
    zero_kernel<<<2048, 256, 0, stream>>>(ws, zTotal);

    fold_kernel<<<1, 64, 0, stream>>>(Wn, bn, We, be, Wm, bm, ws);

    long long eThreads = (long long)nEdges * 32;
    int eBlocks = (int)((eThreads + 255) / 256);
    edge_kernel<<<eBlocks, 256, 0, stream>>>(x, ea, eidx, ws, aggr, nEdges);

    count_kernel<<<256, 256, 0, stream>>>(batc, ws + WS_CNT, nNodes);

    int tiles = nNodes / 16;             // 3125 (exact)
    upd_pool_kernel<<<tiles, 32, 0, stream>>>(aggr, Wupd, bupd, batc, ws + WS_SUMS);

    out_kernel<<<1, 64, 0, stream>>>(ws, Wout, bout, out);
}
